// LSTM_1503238554284
// MI455X (gfx1250) — matrix operand order repaired and checked by reading, not compiled, not run on hardware
//
#include <hip/hip_runtime.h>
#include <hip/hip_bf16.h>
#include <math.h>

// Problem dims
#define B_  64
#define T_  256
#define D_  512
#define H_  512
#define G_  2048           // 4*H
#define M_  (B_*T_)        // 16384

// Scan kernel decomposition
#define JB   32            // h-columns per block
#define NBJ  (H_/JB)       // 16 blocks per direction
#define NBLK (2*NBJ)       // 32 total blocks in the persistent scan grid

// Dynamic LDS layout for the scan kernel
#define WH_LDS_BYTES  (4 * JB * H_ * 2)            // 4 gate chunks x 32 rows x 512 bf16 = 128 KB
#define PRES_LDS_BYTES (B_ * 4 * JB * 4)           // 64 x 128 f32 = 32 KB
#define SCAN_LDS_BYTES (WH_LDS_BYTES + PRES_LDS_BYTES)

typedef __attribute__((ext_vector_type(16))) __bf16 v16bf;
typedef __attribute__((ext_vector_type(8)))  __bf16 v8bf;
typedef __attribute__((ext_vector_type(8)))  float  v8f;
typedef unsigned int u32x4  __attribute__((ext_vector_type(4)));
typedef int          i32x4v __attribute__((ext_vector_type(4)));
typedef int          i32x8v __attribute__((ext_vector_type(8)));

__device__ __forceinline__ unsigned short f2bf(float f) {
  unsigned u = __float_as_uint(f);
  u += 0x7FFFu + ((u >> 16) & 1u);          // round-to-nearest-even
  return (unsigned short)(u >> 16);
}

__device__ __forceinline__ v8bf ld8(const unsigned short* p) {
  return *reinterpret_cast<const v8bf*>(p);
}
__device__ __forceinline__ v16bf ld16(const unsigned short* p) {
  return *reinterpret_cast<const v16bf*>(p);
}
__device__ __forceinline__ v16bf mk16(v8bf lo, v8bf hi) {
  v16bf a;
#pragma unroll
  for (int i = 0; i < 8; ++i) { a[i] = lo[i]; a[i + 8] = hi[i]; }
  return a;
}

// Native V_TANH_F32 on gfx1250 when the builtin exists; sigmoid via tanh.
__device__ __forceinline__ float fast_tanh(float x) {
#if __has_builtin(__builtin_amdgcn_tanh_f32)
  return __builtin_amdgcn_tanh_f32(x);
#elif __has_builtin(__builtin_amdgcn_tanhf)
  return __builtin_amdgcn_tanhf(x);
#else
  return tanhf(x);
#endif
}
__device__ __forceinline__ float sigm(float x) {
  return fmaf(0.5f, fast_tanh(0.5f * x), 0.5f);
}

// Monotonic-counter grid barrier across the 32 persistent blocks.
__device__ __forceinline__ void gbar(unsigned* bar, unsigned target) {
  __threadfence();          // publish this thread's global writes (device scope)
  __syncthreads();
  if (threadIdx.x == 0) {
    __hip_atomic_fetch_add(bar, 1u, __ATOMIC_ACQ_REL, __HIP_MEMORY_SCOPE_AGENT);
    while (__hip_atomic_load(bar, __ATOMIC_ACQUIRE, __HIP_MEMORY_SCOPE_AGENT) < target) {
      __builtin_amdgcn_s_sleep(1);
    }
  }
  __syncthreads();
}

// 1-D contiguous TDM copy: global -> LDS, ndw dwords (ndw < 65536).
// Descriptor bitfields per CDNA5 ISA ch.8 (D# group0/group1).
#if __has_builtin(__builtin_amdgcn_tensor_load_to_lds)
#define HAVE_TDM 1
__device__ __forceinline__ void tdm_load_1d(const void* lds_dst, const void* gsrc,
                                            unsigned ndw) {
  unsigned lds_off = (unsigned)(unsigned long long)lds_dst;   // low 32 bits = LDS offset
  unsigned long long ga = (unsigned long long)gsrc;
  u32x4 g0;
  g0[0] = 1u;                                              // count=1, user descriptor
  g0[1] = lds_off;                                         // lds_addr
  g0[2] = (unsigned)ga;                                    // global_addr[31:0]
  g0[3] = ((unsigned)(ga >> 32) & 0x01FFFFFFu) | (2u << 30);  // global_addr[56:32] | type=2
  i32x8v g1;
  g1[0] = (int)(2u << 16);                                 // data_size = 4 bytes
  g1[1] = (int)((ndw & 0xFFFFu) << 16);                    // tensor_dim0[15:0]
  g1[2] = (int)(((ndw >> 16) & 0xFFFFu) | (1u << 16));     // tensor_dim0[31:16], tensor_dim1=1
  g1[3] = (int)(ndw << 16);                                // tile_dim0 = ndw
  g1[4] = 1;                                               // tile_dim1 = 1
  g1[5] = (int)ndw;                                        // tensor_dim0_stride[31:0]
  g1[6] = 0;
  g1[7] = 0;
  i32x4v gz = {0, 0, 0, 0};
#if defined(__clang_major__) && (__clang_major__ >= 23)
  i32x8v gz8 = {0, 0, 0, 0, 0, 0, 0, 0};
  __builtin_amdgcn_tensor_load_to_lds(g0, g1, gz, gz, gz8, 0);
#else
  __builtin_amdgcn_tensor_load_to_lds(g0, g1, gz, gz, 0);
#endif
}
#else
#define HAVE_TDM 0
#endif

// ---------------------------------------------------------------- f32 -> bf16
__global__ void cvt_bf16_kernel(const float* __restrict__ in,
                                unsigned short* __restrict__ out, int n) {
  int i = blockIdx.x * blockDim.x + threadIdx.x;
  int stride = gridDim.x * blockDim.x;
  for (; i < n; i += stride) out[i] = f2bf(in[i]);
}

// -------------------------------------------- pre = x @ Wi^T + bi  (per dir)
// Block: 256 thr (8 waves) -> 128x128 output tile. Wave w: 16x128 strip.
__global__ void __launch_bounds__(256) input_gemm_kernel(
    const unsigned short* __restrict__ xbf,   // [M_][D_] bf16
    const unsigned short* __restrict__ Wibf,  // [2][G_][D_] bf16 (row-major)
    const float* __restrict__ bi_f, const float* __restrict__ bi_bk,
    float* __restrict__ pre) {                // [2][M_][G_] f32
  const int dir = blockIdx.z;
  const unsigned short* W = Wibf + (size_t)dir * G_ * D_;
  const float* bias = dir ? bi_bk : bi_f;
  float* out = pre + (size_t)dir * M_ * G_;

  const int tid = threadIdx.x;
  const int lane = tid & 31, wave = tid >> 5;
  const int lm = lane & 15, lk = lane >> 4;
  const int m0 = blockIdx.x * 128 + wave * 16;
  const int n0 = blockIdx.y * 128;

  const unsigned short* aptr = xbf + (size_t)(m0 + lm) * D_ + lk * 16;
  const unsigned short* wp[8];
#pragma unroll
  for (int nt = 0; nt < 8; ++nt)
    wp[nt] = W + (size_t)(n0 + nt * 16 + lm) * D_ + lk * 16;

  v8f zero;
#pragma unroll
  for (int i = 0; i < 8; ++i) zero[i] = 0.0f;
  v8f acc[8];
#pragma unroll
  for (int nt = 0; nt < 8; ++nt) acc[nt] = zero;

  for (int k0 = 0; k0 < D_; k0 += 32) {
    v16bf a = ld16(aptr + k0);
#pragma unroll
    for (int nt = 0; nt < 8; ++nt) {
      v16bf b = ld16(wp[nt] + k0);
      acc[nt] = __builtin_amdgcn_wmma_f32_16x16x32_bf16(
          false, a, false, b, (short)0, acc[nt], false, false);
    }
  }

  const int rbase = m0 + lk * 8;
#pragma unroll
  for (int nt = 0; nt < 8; ++nt) {
    int col = n0 + nt * 16 + lm;
    float bv = bias[col];
#pragma unroll
    for (int v = 0; v < 8; ++v)
      out[(size_t)(rbase + v) * G_ + col] = acc[nt][v] + bv;
  }
}

// ----------------------------------------------------- persistent LSTM scan
// Grid: (NBJ, 2). Block owns h-columns [j0, j0+32) of its direction.
// Wh slice (128 KB) staged to LDS ONCE via the Tensor Data Mover; per step the
// block computes its 64x128 gate preacts with WMMA (A from global h_bf, B from
// LDS), exchanges via LDS, applies gates with native v_tanh, publishes h.
__global__ void __launch_bounds__(256) lstm_scan_kernel(
    const float* __restrict__ pre,            // [2][M_][G_] f32
    const unsigned short* __restrict__ Whbf,  // [2][G_][H_] bf16
    const float* __restrict__ bh_f, const float* __restrict__ bh_bk,
    const float* __restrict__ h0, const float* __restrict__ c0,  // [B_][2H]
    const float* __restrict__ mask,           // [B_][T_]
    float* __restrict__ h_state,              // [2][B_][H_] f32
    unsigned short* __restrict__ h_bf,        // [2][B_][H_] bf16
    float* __restrict__ out,                  // h_mat | c_mat
    unsigned* __restrict__ bar) {
  extern __shared__ __align__(128) unsigned char smem[];
  unsigned short* whSm = (unsigned short*)smem;                  // [4*JB][H_] bf16
  float* preS = (float*)(smem + WH_LDS_BYTES);                   // [B_][4*JB] f32

  const int dir = blockIdx.y;
  const int j0 = blockIdx.x * JB;
  const int tid = threadIdx.x;
  const int lane = tid & 31, wave = tid >> 5;
  const int lm = lane & 15, lk = lane >> 4;

  const float* preD = pre + (size_t)dir * M_ * G_;
  const unsigned short* Wh = Whbf + (size_t)dir * G_ * H_;
  const float* bh = dir ? bh_bk : bh_f;
  float* hs = h_state + (size_t)dir * B_ * H_;
  unsigned short* hb = h_bf + (size_t)dir * B_ * H_;
  float* hout = out;
  float* cout = out + (size_t)B_ * T_ * (2 * H_);

  // ---- stage this block's Wh slice into LDS: 4 contiguous 32KB gate chunks
#if HAVE_TDM
  if (wave == 0) {
#pragma unroll
    for (int g = 0; g < 4; ++g)
      tdm_load_1d(whSm + (size_t)g * JB * H_,
                  Wh + (size_t)(g * H_ + j0) * H_,
                  (JB * H_ * 2) / 4 /* = 8192 dwords */);
  }
#else
  for (int g = 0; g < 4; ++g) {
    const uint4* src = (const uint4*)(Wh + (size_t)(g * H_ + j0) * H_);
    uint4* dst = (uint4*)(whSm + (size_t)g * JB * H_);
    for (int i = tid; i < (JB * H_ * 2) / 16; i += 256) dst[i] = src[i];
  }
#endif

  // Elementwise ownership: thread -> column j, rows mb..mb+7; c kept in regs.
  const int j  = tid & (JB - 1);
  const int Jg = j0 + j;
  const int mb = (tid >> 5) * 8;
  float c_reg[8];

  unsigned bar_target = NBLK;

  // ---- init h, c from h0/c0 (overlaps the TDM transfer)
#pragma unroll
  for (int r = 0; r < 8; ++r) {
    int m = mb + r;
    float h0v = h0[(size_t)m * (2 * H_) + dir * H_ + Jg];
    c_reg[r]  = c0[(size_t)m * (2 * H_) + dir * H_ + Jg];
    hs[(size_t)m * H_ + Jg] = h0v;
    hb[(size_t)m * H_ + Jg] = f2bf(h0v);
  }
#if HAVE_TDM
  if (wave == 0) __builtin_amdgcn_s_wait_tensorcnt(0);
#endif
  gbar(bar, bar_target); bar_target += NBLK;   // h/c visible + LDS weights ready

  // ---- wave tile assignment: mt = M tile (4), 4 of 8 N tiles per wave
  const int mt = wave & 3;
  const int ntb = (wave >> 2) * 4;
  const int arow = mt * 16 + lm;
  const unsigned short* aptr = hb + (size_t)arow * H_ + lk * 16;
  const unsigned short* wp[4];   // B fragments now come from LDS
  int colL[4];
#pragma unroll
  for (int nt = 0; nt < 4; ++nt) {
    int ntg = ntb + nt;
    int gate = ntg >> 1, jt = ntg & 1;
    wp[nt] = whSm + (size_t)(gate * JB + jt * 16 + lm) * H_ + lk * 16;
    colL[nt] = gate * JB + jt * 16 + lm;          // LDS preact column
  }

  v8f zero;
#pragma unroll
  for (int i = 0; i < 8; ++i) zero[i] = 0.0f;

  for (int s = 0; s < T_; ++s) {
    const int t = dir ? (T_ - 1 - s) : s;

    // ---- recurrent GEMM: h @ Wh^T for this block's 128 gate columns
    v8f acc[4];
#pragma unroll
    for (int nt = 0; nt < 4; ++nt) acc[nt] = zero;

#pragma unroll 2
    for (int k0 = 0; k0 < H_; k0 += 32) {
      v16bf a = ld16(aptr + k0);
#pragma unroll
      for (int nt = 0; nt < 4; ++nt) {
        v16bf b = ld16(wp[nt] + k0);
        acc[nt] = __builtin_amdgcn_wmma_f32_16x16x32_bf16(
            false, a, false, b, (short)0, acc[nt], false, false);
      }
    }

#pragma unroll
    for (int nt = 0; nt < 4; ++nt) {
      int rbase = mt * 16 + lk * 8;
#pragma unroll
      for (int v = 0; v < 8; ++v)
        preS[(rbase + v) * (4 * JB) + colL[nt]] = acc[nt][v];
    }

    // all blocks finished READING h_t (and this block's LDS preacts complete)
    gbar(bar, bar_target); bar_target += NBLK;

    // ---- gates + cell update + publish h_{t+1}
#pragma unroll
    for (int r = 0; r < 8; ++r) {
      int m = mb + r;
      const float* pr = preD + ((size_t)m * T_ + t) * G_;
      float iv = sigm     (preS[m * (4*JB) + 0*JB + j] + pr[0*H_ + Jg] + bh[0*H_ + Jg]);
      float fv = sigm     (preS[m * (4*JB) + 1*JB + j] + pr[1*H_ + Jg] + bh[1*H_ + Jg]);
      float ov = sigm     (preS[m * (4*JB) + 2*JB + j] + pr[2*H_ + Jg] + bh[2*H_ + Jg]);
      float gv = fast_tanh(preS[m * (4*JB) + 3*JB + j] + pr[3*H_ + Jg] + bh[3*H_ + Jg]);
      float cp = c_reg[r];
      float ct = cp * fv + iv * gv;
      float ht = ov * fast_tanh(ct);
      float mk = mask[(size_t)m * T_ + t];
      float hp = hs[(size_t)m * H_ + Jg];
      float hn = ht * mk + hp * (1.0f - mk);
      float cn = ct * mk + cp * (1.0f - mk);
      c_reg[r] = cn;
      hs[(size_t)m * H_ + Jg] = hn;
      hb[(size_t)m * H_ + Jg] = f2bf(hn);
      size_t oidx = ((size_t)m * T_ + t) * (2 * H_) + dir * H_ + Jg;
      hout[oidx] = hn;
      cout[oidx] = cn;
    }

    // all blocks finished WRITING h_{t+1}
    gbar(bar, bar_target); bar_target += NBLK;
  }
}

// ---------------------------------------------------------------------------
extern "C" void kernel_launch(void* const* d_in, const int* in_sizes, int n_in,
                              void* d_out, int out_size, void* d_ws, size_t ws_size,
                              hipStream_t stream) {
  const float* x    = (const float*)d_in[0];
  const float* h0   = (const float*)d_in[1];
  const float* c0   = (const float*)d_in[2];
  const float* mask = (const float*)d_in[3];
  const float* Wi   = (const float*)d_in[4];
  const float* bi   = (const float*)d_in[5];
  const float* Wh   = (const float*)d_in[6];
  const float* bh   = (const float*)d_in[7];
  const float* Wi_b = (const float*)d_in[8];
  const float* bi_b = (const float*)d_in[9];
  const float* Wh_b = (const float*)d_in[10];
  const float* bh_b = (const float*)d_in[11];
  float* out = (float*)d_out;

  char* ws = (char*)d_ws;
  size_t off = 0;
  unsigned*       bar     = (unsigned*)(ws + off);       off += 256;
  unsigned short* xbf     = (unsigned short*)(ws + off); off += (size_t)M_ * D_ * 2;
  unsigned short* Wibf    = (unsigned short*)(ws + off); off += (size_t)2 * G_ * D_ * 2;
  unsigned short* Whbf    = (unsigned short*)(ws + off); off += (size_t)2 * G_ * H_ * 2;
  float*          pre     = (float*)(ws + off);          off += (size_t)2 * M_ * G_ * 4;
  float*          h_state = (float*)(ws + off);          off += (size_t)2 * B_ * H_ * 4;
  unsigned short* h_bf    = (unsigned short*)(ws + off); off += (size_t)2 * B_ * H_ * 2;

  hipMemsetAsync(bar, 0, 256, stream);   // deterministic barrier state each launch

  // Allow 160 KB dynamic LDS for the scan kernel (WGP has 320 KB).
  hipFuncSetAttribute((const void*)lstm_scan_kernel,
                      hipFuncAttributeMaxDynamicSharedMemorySize, SCAN_LDS_BYTES);

  auto cvt = [&](const float* src, unsigned short* dst, size_t n) {
    int blocks = (int)((n + 255) / 256);
    if (blocks > 4096) blocks = 4096;
    cvt_bf16_kernel<<<blocks, 256, 0, stream>>>(src, dst, (int)n);
  };
  cvt(x,    xbf,                      (size_t)M_ * D_);
  cvt(Wi,   Wibf,                     (size_t)G_ * D_);
  cvt(Wi_b, Wibf + (size_t)G_ * D_,   (size_t)G_ * D_);
  cvt(Wh,   Whbf,                     (size_t)G_ * H_);
  cvt(Wh_b, Whbf + (size_t)G_ * H_,   (size_t)G_ * H_);

  input_gemm_kernel<<<dim3(M_ / 128, G_ / 128, 2), 256, 0, stream>>>(
      xbf, Wibf, bi, bi_b, pre);

  lstm_scan_kernel<<<dim3(NBJ, 2), 256, SCAN_LDS_BYTES, stream>>>(
      pre, Whbf, bh, bh_b, h0, c0, mask, h_state, h_bf, out, bar);
}
